// SimpleMHA_5265629905473
// MI455X (gfx1250) — matrix operand order repaired and checked by reading, not compiled, not run on hardware
//
#include <hip/hip_runtime.h>

// ---------------------------------------------------------------------------
// MHA forward, MI455X (gfx1250, wave32, WMMA, async global->LDS staging).
// Pipeline: cvt fp32->bf16  ->  fused QKV GEMM (WMMA bf16)  ->
//           flash-attention (WMMA bf16, online softmax, no HBM attn matrix) ->
//           output GEMM (WMMA bf16, fp32 epilogue).
// All tile staging uses GLOBAL_LOAD_ASYNC_TO_LDS_B128 (ASYNCcnt) so the
// compute wave never round-trips tiles through VGPRs; double buffering is
// done by waiting ASYNCcnt <= #ops-of-newest-tile (async loads complete
// in order per the CDNA5 ISA).
// ---------------------------------------------------------------------------

#define EMBED 1024
#define HEADS 16
#define HDIM  64
#define SEQ   2048
#define BATCH 2
#define MTOT  (BATCH * SEQ)   // 4096
#define ATTN_SCALE 0.125f     // HDIM^-0.5

typedef __attribute__((ext_vector_type(16))) __bf16 bf16x16;
typedef __attribute__((ext_vector_type(8)))  __bf16 bf16x8;
typedef __attribute__((ext_vector_type(8)))  float  f32x8;

union ABFrag { bf16x16 v; bf16x8 h[2]; };

__device__ __forceinline__ f32x8 wmma_bf16(bf16x16 a, bf16x16 b, f32x8 c) {
  // D = A(16x32) * B(32x16) + C, f32 accumulate
  return __builtin_amdgcn_wmma_f32_16x16x32_bf16(false, a, false, b,
                                                 (short)0, c, false, false);
}

__device__ __forceinline__ void zero8(f32x8& v) {
#pragma unroll
  for (int r = 0; r < 8; ++r) v[r] = 0.0f;
}

// Async DMA of 16B/lane from global to LDS (CDNA5 ASYNCcnt path).
// Generic pointers to __shared__ carry the LDS byte offset in their low 32
// bits (LDS aperture: addr[31:0] = offset), which is what VDST wants.
__device__ __forceinline__ void async_b128(const __bf16* gsrc, __bf16* ldst) {
  unsigned l = (unsigned)(size_t)ldst;
  unsigned long long g = (unsigned long long)(size_t)gsrc;
  asm volatile("global_load_async_to_lds_b128 %0, %1, off"
               :: "v"(l), "v"(g) : "memory");
}

#if __has_builtin(__builtin_amdgcn_s_wait_asynccnt)
#define WAIT_ASYNC(n) __builtin_amdgcn_s_wait_asynccnt(n)
#else
#define WAIT_ASYNC(n) asm volatile("s_wait_asynccnt %0" ::"i"(n) : "memory")
#endif

// A-frag (16x32, row per lane). ISA layout: lane L holds row m=L%16;
// VGPR0..3 = K 8h..8h+7, VGPR4..7 = K 16+8h..16+8h+7  (h = L/16).
__device__ __forceinline__ bf16x16 load_a_frag(const __bf16* tile, int ld,
                                               int mb, int kb) {
  const int lane = threadIdx.x & 31, half = lane >> 4, l15 = lane & 15;
  const __bf16* row = tile + (mb + l15) * ld + kb;
  ABFrag f;
  f.h[0] = *(const bf16x8*)(row + 8 * half);
  f.h[1] = *(const bf16x8*)(row + 16 + 8 * half);
  return f.v;
}

// B-frag (32x16, col per lane) read from an (N x K) row-major tile (B = W^T).
// lane L holds col n=L%16; VGPR0..3 = K 8h..8h+7, VGPR4..7 = K 16+8h..16+8h+7.
__device__ __forceinline__ bf16x16 load_b_frag(const __bf16* tile, int ld,
                                               int nb, int kb) {
  const int lane = threadIdx.x & 31, half = lane >> 4, l15 = lane & 15;
  const __bf16* row = tile + (nb + l15) * ld + kb;
  ABFrag f;
  f.h[0] = *(const bf16x8*)(row + 8 * half);
  f.h[1] = *(const bf16x8*)(row + 16 + 8 * half);
  return f.v;
}

// half-wave (16 lane) row reductions; wave32 shuffles stay in-half for mask<16
__device__ __forceinline__ float rmax16(float v) {
#pragma unroll
  for (int m = 1; m < 16; m <<= 1) v = fmaxf(v, __shfl_xor(v, m, 32));
  return v;
}
__device__ __forceinline__ float rsum16(float v) {
#pragma unroll
  for (int m = 1; m < 16; m <<= 1) v += __shfl_xor(v, m, 32);
  return v;
}

// ---------------------------------------------------------------------------
__global__ void cvt_f32_bf16_kernel(const float* __restrict__ src,
                                    __bf16* __restrict__ dst, int n4) {
  int i = blockIdx.x * blockDim.x + threadIdx.x;
  if (i < n4) {
    float4 v = ((const float4*)src)[i];
    union { __bf16 b[4]; uint2 u; } o;
    o.b[0] = (__bf16)v.x; o.b[1] = (__bf16)v.y;
    o.b[2] = (__bf16)v.z; o.b[3] = (__bf16)v.w;
    ((uint2*)dst)[i] = o.u;
  }
}

// ---------------------------------------------------------------------------
// Fused QKV projection: y = x @ W^T + b, scattered to attention layouts.
// grid (E/64, M/64, 3); block 128 (4 waves); 64x64 tile, double-buffered
// async LDS staging (4 async B128/wave/tile-pair).
// z==0 -> Q (B,H,N,D) bf16 ; z==1 -> K (B,H,N,D) ; z==2 -> V^T (B,H,D,N)
// ---------------------------------------------------------------------------
#define GLDS 40  // padded LDS row stride (elems) for 64x32 tiles

__global__ __launch_bounds__(128) void gemm_qkv_kernel(
    const __bf16* __restrict__ X,
    const __bf16* __restrict__ Wq, const __bf16* __restrict__ Wk,
    const __bf16* __restrict__ Wv,
    const float* __restrict__ bq, const float* __restrict__ bk,
    const float* __restrict__ bv,
    __bf16* __restrict__ Qo, __bf16* __restrict__ Ko,
    __bf16* __restrict__ Vto) {
  __shared__ __bf16 lx[2][64 * GLDS];
  __shared__ __bf16 lw[2][64 * GLDS];

  const int nb = blockIdx.x * 64;
  const int mb = blockIdx.y * 64;
  const int z  = blockIdx.z;
  const __bf16* W   = (z == 0) ? Wq : ((z == 1) ? Wk : Wv);
  const float* bias = (z == 0) ? bq : ((z == 1) ? bk : bv);
  __bf16* out       = (z == 0) ? Qo : ((z == 1) ? Ko : Vto);

  const int w = threadIdx.x >> 5, lane = threadIdx.x & 31;
  const int half = lane >> 4, l15 = lane & 15;
  const int moff = (w >> 1) * 32, noff = (w & 1) * 32;

  f32x8 acc[2][2];
#pragma unroll
  for (int i = 0; i < 2; ++i)
#pragma unroll
    for (int j = 0; j < 2; ++j) zero8(acc[i][j]);

  // each thread DMAs 32B of x-tile + 32B of w-tile = 4 async ops / tile pair
  auto copyTiles = [&](int buf, int kstep) {
    const int t = threadIdx.x, row = t >> 1, part = t & 1;
    const __bf16* sx = X + (size_t)(mb + row) * EMBED + kstep * 32 + part * 16;
    __bf16* dx = &lx[buf][row * GLDS + part * 16];
    async_b128(sx, dx);
    async_b128(sx + 8, dx + 8);
    const __bf16* sw = W + (size_t)(nb + row) * EMBED + kstep * 32 + part * 16;
    __bf16* dw = &lw[buf][row * GLDS + part * 16];
    async_b128(sw, dw);
    async_b128(sw + 8, dw + 8);
  };

  copyTiles(0, 0);
  const int NK = EMBED / 32;
  for (int kb = 0; kb < NK; ++kb) {
    const int cur = kb & 1;
    if (kb + 1 < NK) {
      copyTiles(cur ^ 1, kb + 1);
      WAIT_ASYNC(4);  // newest 4 = next tile still in flight; cur complete
    } else {
      WAIT_ASYNC(0);
    }
    __syncthreads();  // all waves' async writes to cur now LDS-visible
    bf16x16 a0 = load_a_frag(lx[cur], GLDS, moff, 0);
    bf16x16 a1 = load_a_frag(lx[cur], GLDS, moff + 16, 0);
    bf16x16 b0 = load_b_frag(lw[cur], GLDS, noff, 0);
    bf16x16 b1 = load_b_frag(lw[cur], GLDS, noff + 16, 0);
    acc[0][0] = wmma_bf16(a0, b0, acc[0][0]);
    acc[0][1] = wmma_bf16(a0, b1, acc[0][1]);
    acc[1][0] = wmma_bf16(a1, b0, acc[1][0]);
    acc[1][1] = wmma_bf16(a1, b1, acc[1][1]);
    __syncthreads();  // reads of cur done before next iter overwrites it
  }

#pragma unroll
  for (int i = 0; i < 2; ++i)
#pragma unroll
    for (int j = 0; j < 2; ++j) {
      const int e = nb + noff + 16 * j + l15;  // output feature
      const float bb = bias[e];
      const int hh = e >> 6, dd = e & 63;
#pragma unroll
      for (int r = 0; r < 8; ++r) {
        const int m = mb + moff + 16 * i + r + 8 * half;  // global row
        const int bi = m >> 11, nn = m & (SEQ - 1);
        const float val = acc[i][j][r] + bb;
        if (z != 2)
          out[(((size_t)(bi * HEADS + hh)) * SEQ + nn) * HDIM + dd] =
              (__bf16)val;
        else
          out[(((size_t)(bi * HEADS + hh)) * HDIM + dd) * SEQ + nn] =
              (__bf16)val;
      }
    }
}

// ---------------------------------------------------------------------------
// Flash attention: grid (SEQ/64, B*H); block 128 = 4 waves.
// Wave w owns query rows w*16..w*16+15 of this 64-row block; O, m, l live in
// VGPRs; K/V^T tiles double-buffered in LDS via async DMA; P transposed
// C->A layout via per-wave LDS scratch.
// ---------------------------------------------------------------------------
#define ALDS 72  // padded stride for 64-wide attention tiles

__global__ __launch_bounds__(128) void attn_kernel(
    const __bf16* __restrict__ Q, const __bf16* __restrict__ K,
    const __bf16* __restrict__ Vt, __bf16* __restrict__ Ctx) {
  __shared__ __bf16 lk[2][64 * ALDS];   // 64 keys x 64 d
  __shared__ __bf16 lv[2][64 * ALDS];   // 64 d    x 64 keys (V^T)
  __shared__ __bf16 lp[4][16 * ALDS];   // per-wave P scratch 16 x 64

  const int qb = blockIdx.x;   // query block
  const int bh = blockIdx.y;   // b*HEADS + h
  const int w = threadIdx.x >> 5, lane = threadIdx.x & 31;
  const int half = lane >> 4, l15 = lane & 15;

  const __bf16* qbase = Q + ((size_t)bh * SEQ + qb * 64) * HDIM;
  const __bf16* kbase = K + (size_t)bh * SEQ * HDIM;
  const __bf16* vbase = Vt + (size_t)bh * HDIM * SEQ;

  // each thread DMAs 64B of K-tile + 64B of V-tile = 8 async ops / tile pair
  auto copyTiles = [&](int buf, int kb) {
    const int t = threadIdx.x, row = t >> 1, part = t & 1;
    const __bf16* sk = kbase + (size_t)(kb * 64 + row) * HDIM + part * 32;
    __bf16* dk = &lk[buf][row * ALDS + part * 32];
    const __bf16* sv = vbase + (size_t)row * SEQ + kb * 64 + part * 32;
    __bf16* dv = &lv[buf][row * ALDS + part * 32];
#pragma unroll
    for (int i = 0; i < 4; ++i) {
      async_b128(sk + i * 8, dk + i * 8);
      async_b128(sv + i * 8, dv + i * 8);
    }
  };

  // Q A-frags, loaded once (rows w*16..+15, K-dim = d in chunks of 32)
  const __bf16* qrow = qbase + (size_t)(w * 16 + l15) * HDIM;
  ABFrag qf[2];
#pragma unroll
  for (int ks = 0; ks < 2; ++ks) {
    qf[ks].h[0] = *(const bf16x8*)(qrow + 32 * ks + 8 * half);
    qf[ks].h[1] = *(const bf16x8*)(qrow + 32 * ks + 16 + 8 * half);
  }

  f32x8 o[4];
#pragma unroll
  for (int dt = 0; dt < 4; ++dt) zero8(o[dt]);
  float mrow[8], lrow[8];
#pragma unroll
  for (int r = 0; r < 8; ++r) { mrow[r] = -1e30f; lrow[r] = 0.0f; }

  copyTiles(0, 0);
  const int NT = SEQ / 64;
  for (int kb = 0; kb < NT; ++kb) {
    const int cur = kb & 1;
    if (kb + 1 < NT) {
      copyTiles(cur ^ 1, kb + 1);
      WAIT_ASYNC(8);  // newest 8 = next tile pair; cur tiles complete
    } else {
      WAIT_ASYNC(0);
    }
    __syncthreads();

    // S = (q @ k^T) * scale : 4 n-tiles x 2 k-steps
    f32x8 sf[4];
#pragma unroll
    for (int nt = 0; nt < 4; ++nt) {
      f32x8 s; zero8(s);
#pragma unroll
      for (int ks = 0; ks < 2; ++ks) {
        bf16x16 bk16 = load_b_frag(lk[cur], ALDS, nt * 16, 32 * ks);
        s = wmma_bf16(qf[ks].v, bk16, s);
      }
#pragma unroll
      for (int r = 0; r < 8; ++r) s[r] *= ATTN_SCALE;
      sf[nt] = s;
    }

    // online softmax update
    float mnew[8];
#pragma unroll
    for (int r = 0; r < 8; ++r) {
      float v = fmaxf(fmaxf(sf[0][r], sf[1][r]), fmaxf(sf[2][r], sf[3][r]));
      v = rmax16(v);
      mnew[r] = fmaxf(mrow[r], v);
      const float alpha = __expf(mrow[r] - mnew[r]);
      lrow[r] *= alpha;
#pragma unroll
      for (int dt = 0; dt < 4; ++dt) o[dt][r] *= alpha;
      mrow[r] = mnew[r];
    }

    // P = exp(S - m); spill to per-wave LDS for C->A layout change
#pragma unroll
    for (int nt = 0; nt < 4; ++nt)
#pragma unroll
      for (int r = 0; r < 8; ++r) {
        const float p = __expf(sf[nt][r] - mnew[r]);
        sf[nt][r] = p;
        lp[w][(r + 8 * half) * ALDS + nt * 16 + l15] = (__bf16)p;
      }
#pragma unroll
    for (int r = 0; r < 8; ++r) {
      float rs = sf[0][r] + sf[1][r] + sf[2][r] + sf[3][r];
      lrow[r] += rsum16(rs);
    }

    // O += P @ V : 2 k-steps (keys) x 4 d-tiles
#pragma unroll
    for (int ks = 0; ks < 2; ++ks) {
      bf16x16 a = load_a_frag(lp[w], ALDS, 0, 32 * ks);
#pragma unroll
      for (int dt = 0; dt < 4; ++dt) {
        bf16x16 bv16 = load_b_frag(lv[cur], ALDS, dt * 16, 32 * ks);
        o[dt] = wmma_bf16(a, bv16, o[dt]);
      }
    }
    __syncthreads();  // reads of cur done before next iter overwrites it
  }

  // normalize and write ctx (B, N, E) bf16
  const int b = bh >> 4, h = bh & 15;
#pragma unroll
  for (int r = 0; r < 8; ++r) {
    const float inv = 1.0f / lrow[r];
    const int g = b * SEQ + qb * 64 + w * 16 + r + 8 * half;
#pragma unroll
    for (int dt = 0; dt < 4; ++dt)
      Ctx[(size_t)g * EMBED + h * HDIM + dt * 16 + l15] =
          (__bf16)(o[dt][r] * inv);
  }
}

// ---------------------------------------------------------------------------
// Output projection: out = ctx @ Wo^T + bo, fp32 row-major epilogue.
// ---------------------------------------------------------------------------
__global__ __launch_bounds__(128) void gemm_out_kernel(
    const __bf16* __restrict__ X, const __bf16* __restrict__ W,
    const float* __restrict__ bias, float* __restrict__ Out) {
  __shared__ __bf16 lx[2][64 * GLDS];
  __shared__ __bf16 lw[2][64 * GLDS];

  const int nb = blockIdx.x * 64;
  const int mb = blockIdx.y * 64;
  const int w = threadIdx.x >> 5, lane = threadIdx.x & 31;
  const int half = lane >> 4, l15 = lane & 15;
  const int moff = (w >> 1) * 32, noff = (w & 1) * 32;

  f32x8 acc[2][2];
#pragma unroll
  for (int i = 0; i < 2; ++i)
#pragma unroll
    for (int j = 0; j < 2; ++j) zero8(acc[i][j]);

  auto copyTiles = [&](int buf, int kstep) {
    const int t = threadIdx.x, row = t >> 1, part = t & 1;
    const __bf16* sx = X + (size_t)(mb + row) * EMBED + kstep * 32 + part * 16;
    __bf16* dx = &lx[buf][row * GLDS + part * 16];
    async_b128(sx, dx);
    async_b128(sx + 8, dx + 8);
    const __bf16* sw = W + (size_t)(nb + row) * EMBED + kstep * 32 + part * 16;
    __bf16* dw = &lw[buf][row * GLDS + part * 16];
    async_b128(sw, dw);
    async_b128(sw + 8, dw + 8);
  };

  copyTiles(0, 0);
  const int NK = EMBED / 32;
  for (int kb = 0; kb < NK; ++kb) {
    const int cur = kb & 1;
    if (kb + 1 < NK) {
      copyTiles(cur ^ 1, kb + 1);
      WAIT_ASYNC(4);
    } else {
      WAIT_ASYNC(0);
    }
    __syncthreads();
    bf16x16 a0 = load_a_frag(lx[cur], GLDS, moff, 0);
    bf16x16 a1 = load_a_frag(lx[cur], GLDS, moff + 16, 0);
    bf16x16 b0 = load_b_frag(lw[cur], GLDS, noff, 0);
    bf16x16 b1 = load_b_frag(lw[cur], GLDS, noff + 16, 0);
    acc[0][0] = wmma_bf16(a0, b0, acc[0][0]);
    acc[0][1] = wmma_bf16(a0, b1, acc[0][1]);
    acc[1][0] = wmma_bf16(a1, b0, acc[1][0]);
    acc[1][1] = wmma_bf16(a1, b1, acc[1][1]);
    __syncthreads();
  }

#pragma unroll
  for (int i = 0; i < 2; ++i)
#pragma unroll
    for (int j = 0; j < 2; ++j) {
      const int e = nb + noff + 16 * j + l15;
      const float bb = bias[e];
#pragma unroll
      for (int r = 0; r < 8; ++r) {
        const int m = mb + moff + 16 * i + r + 8 * half;
        Out[(size_t)m * EMBED + e] = acc[i][j][r] + bb;
      }
    }
}

// ---------------------------------------------------------------------------
extern "C" void kernel_launch(void* const* d_in, const int* in_sizes, int n_in,
                              void* d_out, int out_size, void* d_ws,
                              size_t ws_size, hipStream_t stream) {
  (void)in_sizes; (void)n_in; (void)out_size; (void)ws_size;
  const float* x  = (const float*)d_in[0];
  const float* Wq = (const float*)d_in[1];
  const float* bq = (const float*)d_in[2];
  const float* Wk = (const float*)d_in[3];
  const float* bk = (const float*)d_in[4];
  const float* Wv = (const float*)d_in[5];
  const float* bv = (const float*)d_in[6];
  const float* Wo = (const float*)d_in[7];
  const float* bo = (const float*)d_in[8];
  float* out = (float*)d_out;

  char* ws = (char*)d_ws;
  const size_t MB = 1024 * 1024;
  __bf16* xb  = (__bf16*)(ws + 0 * MB);    // 8 MB  (4096x1024)
  __bf16* wqb = (__bf16*)(ws + 8 * MB);    // 2 MB each
  __bf16* wkb = (__bf16*)(ws + 10 * MB);
  __bf16* wvb = (__bf16*)(ws + 12 * MB);
  __bf16* wob = (__bf16*)(ws + 14 * MB);
  __bf16* qb  = (__bf16*)(ws + 16 * MB);   // (B,H,N,D) 8 MB
  __bf16* kbf = (__bf16*)(ws + 24 * MB);   // (B,H,N,D) 8 MB
  __bf16* vtb = (__bf16*)(ws + 32 * MB);   // (B,H,D,N) 8 MB
  __bf16* ctx = (__bf16*)(ws + 40 * MB);   // (B,N,E)   8 MB

  cvt_f32_bf16_kernel<<<4096, 256, 0, stream>>>(x, xb, MTOT * EMBED / 4);
  cvt_f32_bf16_kernel<<<1024, 256, 0, stream>>>(Wq, wqb, EMBED * EMBED / 4);
  cvt_f32_bf16_kernel<<<1024, 256, 0, stream>>>(Wk, wkb, EMBED * EMBED / 4);
  cvt_f32_bf16_kernel<<<1024, 256, 0, stream>>>(Wv, wvb, EMBED * EMBED / 4);
  cvt_f32_bf16_kernel<<<1024, 256, 0, stream>>>(Wo, wob, EMBED * EMBED / 4);

  dim3 gp(EMBED / 64, MTOT / 64, 3);
  gemm_qkv_kernel<<<gp, 128, 0, stream>>>(xb, wqb, wkb, wvb, bq, bk, bv, qb,
                                          kbf, vtb);

  attn_kernel<<<dim3(SEQ / 64, BATCH * HEADS), 128, 0, stream>>>(qb, kbf, vtb,
                                                                 ctx);

  gemm_out_kernel<<<dim3(EMBED / 64, MTOT / 64), 128, 0, stream>>>(ctx, wob,
                                                                   bo, out);
}
